// GCN_47656957116627
// MI455X (gfx1250) — compile-verified, matrix operand order checked by reading, not run on hardware
//
#include <hip/hip_runtime.h>
#include <hip/hip_bf16.h>
#include <stdint.h>

#define NNODES 100000
#define NEDGES 1600000
#define FDIM   256
#define HDIM   128
#define H2DIM  64
#define BNEPS  1e-5f

typedef __attribute__((ext_vector_type(16))) __bf16 v16bf;
typedef __attribute__((ext_vector_type(8)))  __bf16 v8bf;
typedef __attribute__((ext_vector_type(8)))  float  v8f;

__device__ __forceinline__ void atomAddF(float* p, float v) {
    // relaxed, agent scope -> global_atomic_add_f32 (no return)
    __hip_atomic_fetch_add(p, v, __ATOMIC_RELAXED, __HIP_MEMORY_SCOPE_AGENT);
}

// ---------------------------------------------------------------- degree/norm
__global__ void fill_ones(float* __restrict__ p, int n) {
    int i = blockIdx.x * blockDim.x + threadIdx.x;
    if (i < n) p[i] = 1.0f;   // self-loop contributes 1 to every node's degree
}

__global__ void deg_count(const long long* __restrict__ dst, float* __restrict__ deg, int e) {
    int i = blockIdx.x * blockDim.x + threadIdx.x;
    if (i < e) atomAddF(&deg[(size_t)dst[i]], 1.0f);
}

__global__ void deg_to_dinv(float* __restrict__ d, int n) {
    int i = blockIdx.x * blockDim.x + threadIdx.x;
    if (i < n) { float v = d[i]; d[i] = (v > 0.f) ? rsqrtf(v) : 0.f; }
}

__global__ void edge_norm(const long long* __restrict__ src, const long long* __restrict__ dst,
                          const float* __restrict__ dinv, float* __restrict__ norm, int e) {
    int i = blockIdx.x * blockDim.x + threadIdx.x;
    if (i < e) norm[i] = dinv[(size_t)src[i]] * dinv[(size_t)dst[i]];
}

// ---------------------------------------------------------------- conversions
__global__ void f32_to_bf16(const float* __restrict__ in, __bf16* __restrict__ out, long long n) {
    long long i = (long long)blockIdx.x * blockDim.x + threadIdx.x;
    if (i < n) out[i] = (__bf16)in[i];
}

// Pack W[K,Nout] (f32 row-major) into bf16 WMMA B-fragments:
// fragment (ks, nt) = 512 bf16: lane-major, each lane's 16 values contiguous.
// Lane l (col = nt*16 + (l&15), koff = (l>>4)*8):
//   elems 0..7  -> K = ks*32 + koff + i
//   elems 8..15 -> K = ks*32 + 16 + koff + (i-8)
__global__ void pack_w(const float* __restrict__ W, __bf16* __restrict__ Wp, int K, int Nout) {
    int idx = blockIdx.x * blockDim.x + threadIdx.x;
    int total = K * Nout;
    if (idx >= total) return;
    int frag = idx >> 9;          // 512 elements per fragment
    int e    = idx & 511;
    int lane = e >> 4;
    int i    = e & 15;
    int tilesN = Nout >> 4;
    int ks = frag / tilesN;
    int nt = frag % tilesN;
    int koff = (lane >> 4) * 8;
    int k = ks * 32 + (i < 8 ? koff + i : 16 + koff + (i - 8));
    int col = nt * 16 + (lane & 15);
    Wp[idx] = (__bf16)W[(size_t)k * Nout + col];
}

// ---------------------------------------------------------------- WMMA GEMM
// C[M, NTILES*16] = A[M,K](bf16 row-major) @ Wp(prepacked bf16 fragments).
// Whole packed weight array staged in LDS once per workgroup; one wave per
// 16-row M-tile, A fragment reused across all NTILES output tiles.
// Epilogue writes both h (gather source) and agg-init h*dinv^2 (self-loop).
template <int K, int NTILES>
__global__ void __launch_bounds__(256)
gemm_wmma_bf16(const __bf16* __restrict__ A, const __bf16* __restrict__ Wp,
               const float* __restrict__ dinv,
               float* __restrict__ H, float* __restrict__ AGG, int tilesM) {
    constexpr int KSTEPS = K / 32;
    constexpr int TOTAL  = KSTEPS * NTILES * 512;      // bf16 elements
    __shared__ __bf16 sW[TOTAL];

    // cooperative stage of packed weights into LDS (16B chunks)
    for (int i = threadIdx.x * 8; i < TOTAL; i += 256 * 8)
        *(v8bf*)(sW + i) = *(const v8bf*)(Wp + i);
    __syncthreads();

    int wid = blockIdx.x * (blockDim.x >> 5) + (threadIdx.x >> 5);
    if (wid >= tilesM) return;                   // wave-uniform: EXEC stays all-1s
    const int lane = threadIdx.x & 31;
    const int l15  = lane & 15;
    const int half = lane >> 4;                  // K-half select per ISA layout
    const int koff = half * 8;
    const int Nout = NTILES * 16;

    const __bf16* arow  = A  + (size_t)(wid * 16 + l15) * K;
    const __bf16* slane = sW + lane * 16;

    v8f acc[NTILES];
    #pragma unroll
    for (int nt = 0; nt < NTILES; ++nt) acc[nt] = (v8f){};

    #pragma unroll 2
    for (int ks = 0; ks < KSTEPS; ++ks) {
        // A fragment: two contiguous 16B bf16 runs from global (streaming)
        v8bf alo = *(const v8bf*)(arow + ks * 32 + koff);
        v8bf ahi = *(const v8bf*)(arow + ks * 32 + 16 + koff);
        v16bf av = __builtin_shufflevector(alo, ahi,
                     0,1,2,3,4,5,6,7,8,9,10,11,12,13,14,15);
        #pragma unroll
        for (int nt = 0; nt < NTILES; ++nt) {
            // B fragment: one 32B LDS read per lane (ds_load_b128 x2)
            v16bf bv = *(const v16bf*)(slane + (size_t)(ks * NTILES + nt) * 512);
            acc[nt] = __builtin_amdgcn_wmma_f32_16x16x32_bf16(
                          false, av, false, bv, (short)0, acc[nt], false, false);
        }
    }
    // C/D layout: VGPR r -> row wid*16 + r + 8*half, col = nt*16 + l15
    const int rowbase = wid * 16 + half * 8;
    #pragma unroll
    for (int r = 0; r < 8; ++r) {
        float dv = dinv[rowbase + r];
        float s  = dv * dv;                      // self-loop weight dinv^2
        #pragma unroll
        for (int nt = 0; nt < NTILES; ++nt) {
            size_t idx = (size_t)(rowbase + r) * Nout + nt * 16 + l15;
            float hval = acc[nt][r];
            H[idx]   = hval;
            AGG[idx] = hval * s;
        }
    }
}

// ---------------------------------------------------------------- aggregation
// one wave covers one edge's feature row in float4 chunks (coalesced, L2-resident)
__global__ void scatter_edges(const long long* __restrict__ src, const long long* __restrict__ dst,
                              const float* __restrict__ norm, const float* __restrict__ h,
                              float* __restrict__ agg, int e, int hout) {
    int chunks = hout >> 2;
    long long gid = (long long)blockIdx.x * blockDim.x + threadIdx.x;
    long long total = (long long)e * chunks;
    if (gid >= total) return;
    int ed = (int)(gid / chunks);
    int f  = (int)(gid % chunks) * 4;
    long long s = src[ed], d = dst[ed];
    float w = norm[ed];
    float4 hv = *(const float4*)(h + (size_t)s * hout + f);
    float* ap = agg + (size_t)d * hout + f;
    atomAddF(ap + 0, hv.x * w);
    atomAddF(ap + 1, hv.y * w);
    atomAddF(ap + 2, hv.z * w);
    atomAddF(ap + 3, hv.w * w);
}

// fused conv-bias + BatchNorm(inference) + ReLU; emits bf16 (next GEMM operand)
__global__ void bn_relu_bf16(const float* __restrict__ agg, const float* __restrict__ b,
                             const float* __restrict__ g, const float* __restrict__ be,
                             const float* __restrict__ m, const float* __restrict__ v,
                             __bf16* __restrict__ out, int n, int hout) {
    long long i = (long long)blockIdx.x * blockDim.x + threadIdx.x;
    long long total = (long long)n * hout;
    if (i >= total) return;
    int f = (int)(i % hout);
    float val = agg[i] + b[f];
    val = g[f] * (val - m[f]) * rsqrtf(v[f] + BNEPS) + be[f];
    out[i] = (__bf16)(val > 0.f ? val : 0.f);
}

// out[n] = sigmoid(dot(h[n,0:64], Wr) + br), h in bf16
__global__ void readout_bf16(const __bf16* __restrict__ h, const float* __restrict__ Wr,
                             const float* __restrict__ br, float* __restrict__ out, int n) {
    int i = blockIdx.x * blockDim.x + threadIdx.x;
    if (i >= n) return;
    const v8bf* hp = (const v8bf*)(h + (size_t)i * H2DIM);
    float acc = 0.f;
    #pragma unroll
    for (int c = 0; c < H2DIM / 8; ++c) {
        v8bf hv = hp[c];
        #pragma unroll
        for (int j = 0; j < 8; ++j) acc += (float)hv[j] * Wr[c * 8 + j];
    }
    acc += br[0];
    out[i] = 1.f / (1.f + __expf(-acc));
}

// ---------------------------------------------------------------- launch
extern "C" void kernel_launch(void* const* d_in, const int* in_sizes, int n_in,
                              void* d_out, int out_size, void* d_ws, size_t ws_size,
                              hipStream_t stream) {
    const float*     x    = (const float*)d_in[0];
    const long long* ei   = (const long long*)d_in[1];   // int64 in reference
    const long long* esrc = ei;
    const long long* edst = ei + NEDGES;
    const float* W1 = (const float*)d_in[2];  const float* b1 = (const float*)d_in[3];
    const float* g1 = (const float*)d_in[4];  const float* be1 = (const float*)d_in[5];
    const float* m1 = (const float*)d_in[6];  const float* v1 = (const float*)d_in[7];
    const float* W2 = (const float*)d_in[8];  const float* b2 = (const float*)d_in[9];
    const float* g2 = (const float*)d_in[10]; const float* be2 = (const float*)d_in[11];
    const float* m2 = (const float*)d_in[12]; const float* v2 = (const float*)d_in[13];
    const float* W3 = (const float*)d_in[14]; const float* b3 = (const float*)d_in[15];
    const float* g3 = (const float*)d_in[16]; const float* be3 = (const float*)d_in[17];
    const float* m3 = (const float*)d_in[18]; const float* v3 = (const float*)d_in[19];
    const float* Wr = (const float*)d_in[20]; const float* br = (const float*)d_in[21];
    float* out = (float*)d_out;

    char* ws = (char*)d_ws;
    auto align_up = [](size_t v) { return (v + 255) & ~(size_t)255; };
    size_t offDinv = 0;
    size_t offNorm = align_up((size_t)NNODES * 4);
    size_t offA    = align_up(offNorm + (size_t)NEDGES * 4);
    size_t offB    = align_up(offA + (size_t)NNODES * HDIM * 4);
    size_t offX    = align_up(offB + (size_t)NNODES * HDIM * 4);
    size_t offWp   = align_up(offX + (size_t)NNODES * FDIM * 2);
    float*  dinv = (float*)(ws + offDinv);
    float*  norm = (float*)(ws + offNorm);
    float*  bufA = (float*)(ws + offA);   // GEMM output h, f32 (gather source)
    float*  bufB = (float*)(ws + offB);   // aggregation target f32
    __bf16* bufX = (__bf16*)(ws + offX);  // bf16 activations (GEMM A operand)
    __bf16* Wp   = (__bf16*)(ws + offWp); // prepacked bf16 weight fragments

    const int T = 256;
    fill_ones  <<<(NNODES + T - 1) / T, T, 0, stream>>>(dinv, NNODES);
    deg_count  <<<(NEDGES + T - 1) / T, T, 0, stream>>>(edst, dinv, NEDGES);
    deg_to_dinv<<<(NNODES + T - 1) / T, T, 0, stream>>>(dinv, NNODES);
    edge_norm  <<<(NEDGES + T - 1) / T, T, 0, stream>>>(esrc, edst, dinv, norm, NEDGES);

    // stage x as bf16
    long long nx = (long long)NNODES * FDIM;
    f32_to_bf16<<<(int)((nx + T - 1) / T), T, 0, stream>>>(x, bufX, nx);

    const int tilesM = NNODES / 16;
    const int gemmBlocks = (tilesM + (T / 32) - 1) / (T / 32);

    auto post_gemm = [&](int Hout, const float* bb, const float* gg, const float* bbe,
                         const float* mm, const float* vv) {
        long long t2 = (long long)NEDGES * (Hout / 4);
        scatter_edges<<<(int)((t2 + T - 1) / T), T, 0, stream>>>(esrc, edst, norm, bufA, bufB,
                                                                 NEDGES, Hout);
        long long t3 = (long long)NNODES * Hout;
        bn_relu_bf16<<<(int)((t3 + T - 1) / T), T, 0, stream>>>(bufB, bb, gg, bbe, mm, vv,
                                                                bufX, NNODES, Hout);
    };

    // Layer 1: K=256, Hout=128 (8 N-tiles, 64KB LDS weight stage)
    pack_w<<<(FDIM * HDIM + T - 1) / T, T, 0, stream>>>(W1, Wp, FDIM, HDIM);
    gemm_wmma_bf16<FDIM, 8><<<gemmBlocks, T, 0, stream>>>(bufX, Wp, dinv, bufA, bufB, tilesM);
    post_gemm(HDIM, b1, g1, be1, m1, v1);

    // Layer 2: K=128, Hout=128 (32KB LDS stage)
    pack_w<<<(HDIM * HDIM + T - 1) / T, T, 0, stream>>>(W2, Wp, HDIM, HDIM);
    gemm_wmma_bf16<HDIM, 8><<<gemmBlocks, T, 0, stream>>>(bufX, Wp, dinv, bufA, bufB, tilesM);
    post_gemm(HDIM, b2, g2, be2, m2, v2);

    // Layer 3: K=128, Hout=64 (4 N-tiles, 16KB LDS stage)
    pack_w<<<(HDIM * H2DIM + T - 1) / T, T, 0, stream>>>(W3, Wp, HDIM, H2DIM);
    gemm_wmma_bf16<HDIM, 4><<<gemmBlocks, T, 0, stream>>>(bufX, Wp, dinv, bufA, bufB, tilesM);
    post_gemm(H2DIM, b3, g3, be3, m3, v3);

    readout_bf16<<<(NNODES + T - 1) / T, T, 0, stream>>>(bufX, Wr, br, out, NNODES);

    (void)in_sizes; (void)n_in; (void)out_size; (void)ws_size;
}
